// iLoRAModel_26551487823968
// MI455X (gfx1250) — matrix operand order repaired and checked by reading, compiled, not run on hardware
//
#include <hip/hip_runtime.h>
#include <hip/hip_bf16.h>
#include <cstdint>
#include <cstddef>

// ------------------- model dimensions (match reference) -------------------
constexpr int B_   = 8;
constexpr int SI_  = 50;
constexpr int SL_  = 256;
constexpr int D_   = 1024;
constexpr int NH_  = 16;
constexpr int DFF_ = 4096;
constexpr int NL_  = 2;
constexpr int V_   = 32000;
constexpr int E_   = 8;
constexpr int R_   = 16;
constexpr float LORA_SCALE_ = 2.0f;   // 32/16
constexpr int HID_ = 128;
constexpr int EH_  = 4;
constexpr int EFF_ = 2048;
constexpr int NI_  = 5000;
constexpr int DH_  = D_ / NH_;    // 64
constexpr int EDH_ = HID_ / EH_;  // 32
constexpr int BSL_ = B_ * SL_;    // 2048

typedef __attribute__((ext_vector_type(16))) _Float16 v16h;
typedef __attribute__((ext_vector_type(8)))  _Float16 v8h;
typedef __attribute__((ext_vector_type(8)))  float    v8f;

__device__ __forceinline__ float gelu_f(float x) {
  float x3 = x * x * x;
  return 0.5f * x * (1.0f + tanhf(0.7978845608028654f * (x + 0.044715f * x3)));
}

// =====================================================================
//  WMMA GEMM with LDS block-tiling + CDNA5 async global->LDS staging.
//  C[M,N] = A[M,K] * W[N,K]^T  (A,W f16 row-major, C f32)
//  Block (256 thr / 8 waves) owns a 128x128 output tile.
//  Per K-chunk (64): stage A-panel 128x64 and B-panel 128x64 into LDS via
//  GLOBAL_LOAD_ASYNC_TO_LDS_B128 (ASYNCcnt), then each wave runs
//  2 M-tiles x 4 N-tiles x 2 k-steps = 16 v_wmma_f32_16x16x32_f16.
//  Fragment layouts follow CDNA5 ISA 7.12.2 (wave32).
// =====================================================================
constexpr int KC_   = 64;        // K-chunk staged per iteration
constexpr int LDSK_ = KC_ + 8;   // padded row stride in halves: 144B = 36 banks

__global__ void __launch_bounds__(256)
wmma_gemm_nt(const _Float16* __restrict__ A,
             const _Float16* __restrict__ W,
             float* __restrict__ C,
             int M, int N, int K) {
  __shared__ _Float16 As[128 * LDSK_];
  __shared__ _Float16 Bs[128 * LDSK_];

  const int tid  = threadIdx.x;
  const int lane = tid & 31;
  const int wave = tid >> 5;
  const int nBlk = N >> 7;                       // 128-col tiles per row band
  const int tm0  = (blockIdx.x / nBlk) << 7;
  const int tn0  = (blockIdx.x % nBlk) << 7;

  const int wm   = wave >> 1;                    // 0..3 : 32-row sub-band
  const int wn   = wave & 1;                     // 0..1 : 64-col sub-band
  const int half = lane >> 4;                    // 0: lanes 0-15, 1: 16-31
  const int l15  = lane & 15;

  // staging: thread t copies 32 consecutive halves (4 x b128) of one row
  const int ldRow = tid >> 1;                    // 0..127
  const int ldCol = (tid & 1) * 32;              // 0 or 32

  const _Float16* gA = A + (size_t)(tm0 + ldRow) * K + ldCol;
  const _Float16* gB = W + (size_t)(tn0 + ldRow) * K + ldCol;
  // flat LDS pointer: low 32 bits are the wave-relative LDS byte address
  const uint32_t ldsA = (uint32_t)(uintptr_t)(As + ldRow * LDSK_ + ldCol);
  const uint32_t ldsB = (uint32_t)(uintptr_t)(Bs + ldRow * LDSK_ + ldCol);

  v8f acc[2][4];
  const v8f vzero = {0.f, 0.f, 0.f, 0.f, 0.f, 0.f, 0.f, 0.f};
#pragma unroll
  for (int mi = 0; mi < 2; ++mi)
#pragma unroll
    for (int j = 0; j < 4; ++j) acc[mi][j] = vzero;

  for (int kc = 0; kc < K; kc += KC_) {
    // ---- async stage A/B panels into LDS (tracked by ASYNCcnt) ----
#pragma unroll
    for (int i = 0; i < 4; ++i) {
      asm volatile("global_load_async_to_lds_b128 %0, %1, off"
                   :: "v"(ldsA + 16u * i), "v"(gA + kc + 8 * i) : "memory");
      asm volatile("global_load_async_to_lds_b128 %0, %1, off"
                   :: "v"(ldsB + 16u * i), "v"(gB + kc + 8 * i) : "memory");
    }
    asm volatile("s_wait_asynccnt 0x0" ::: "memory");
    __syncthreads();

#pragma unroll
    for (int ks = 0; ks < KC_; ks += 32) {
      // A 16x32 fragment: elems 0..7 -> K = ks+8*half+i, 8..15 -> +16
      v16h af[2];
#pragma unroll
      for (int mi = 0; mi < 2; ++mi) {
        const _Float16* ap =
            As + (wm * 32 + mi * 16 + l15) * LDSK_ + ks + half * 8;
        v8h a0 = *(const v8h*)ap;
        v8h a1 = *(const v8h*)(ap + 16);
#pragma unroll
        for (int i = 0; i < 8; ++i) { af[mi][i] = a0[i]; af[mi][i + 8] = a1[i]; }
      }
#pragma unroll
      for (int j = 0; j < 4; ++j) {
        // B 32x16 fragment: elems 0..7 -> K = ks+8*half+i, 8..15 -> +16
        const _Float16* bp =
            Bs + (wn * 64 + j * 16 + l15) * LDSK_ + ks + half * 8;
        v8h b0 = *(const v8h*)bp;
        v8h b1 = *(const v8h*)(bp + 16);
        v16h bf;
#pragma unroll
        for (int i = 0; i < 8; ++i) { bf[i] = b0[i]; bf[i + 8] = b1[i]; }
#pragma unroll
        for (int mi = 0; mi < 2; ++mi)
          acc[mi][j] = __builtin_amdgcn_wmma_f32_16x16x32_f16(
              false, af[mi], false, bf, (short)0, acc[mi][j], false, false);
      }
    }
    __syncthreads();   // all reads done before next chunk overwrites LDS
  }

  // C/D layout: VGPR r -> row +r+8*half, col = l15
#pragma unroll
  for (int mi = 0; mi < 2; ++mi)
#pragma unroll
    for (int j = 0; j < 4; ++j)
#pragma unroll
      for (int r = 0; r < 8; ++r)
        C[(size_t)(tm0 + wm * 32 + mi * 16 + r + half * 8) * N +
          tn0 + wn * 64 + j * 16 + l15] = acc[mi][j][r];
}

// ------------------- elementwise / utility kernels -------------------
__global__ void cast_f32_f16(const float* __restrict__ src,
                             _Float16* __restrict__ dst, int n) {
  int i = blockIdx.x * blockDim.x + threadIdx.x;
  if (i < n) dst[i] = (_Float16)src[i];
}

__global__ void add_inplace_k(float* __restrict__ dst,
                              const float* __restrict__ src, int n) {
  int i = blockIdx.x * blockDim.x + threadIdx.x;
  if (i < n) dst[i] += src[i];
}

// act: 0 = none, 1 = relu, 2 = gelu
__global__ void bias_act_k(float* __restrict__ buf,
                           const float* __restrict__ bias, int M, int N,
                           int act) {
  int i = blockIdx.x * blockDim.x + threadIdx.x;
  if (i >= M * N) return;
  float v = buf[i] + bias[i % N];
  if (act == 1) v = fmaxf(v, 0.f);
  else if (act == 2) v = gelu_f(v);
  buf[i] = v;
}

__global__ void gather_rows_k(const float* __restrict__ table,
                              const int* __restrict__ idx,
                              float* __restrict__ dst, int nrows, int cols) {
  int i = blockIdx.x * blockDim.x + threadIdx.x;
  if (i >= nrows * cols) return;
  int r = i / cols, c = i % cols;
  dst[i] = table[(size_t)idx[r] * cols + c];
}

// out[m,n] = act(X[m,:] . W[n,:] + bias[n])   (small f32 GEMM for encoder)
__global__ void sgemm_f32_k(const float* __restrict__ X,
                            const float* __restrict__ W,
                            const float* __restrict__ bias,
                            float* __restrict__ out, int M, int N, int K,
                            int act) {
  int i = blockIdx.x * blockDim.x + threadIdx.x;
  if (i >= M * N) return;
  int m = i / N, n = i % N;
  const float* xr = X + (size_t)m * K;
  const float* wr = W + (size_t)n * K;
  float acc = bias ? bias[n] : 0.f;
  for (int k = 0; k < K; ++k) acc += xr[k] * wr[k];
  if (act == 1) acc = fmaxf(acc, 0.f);
  else if (act == 2) acc = gelu_f(acc);
  out[i] = acc;
}

// LayerNorm over rows (one block per row); out may alias x.
__global__ void ln_rows_k(const float* __restrict__ x,
                          const float* __restrict__ g,
                          const float* __restrict__ b,
                          float* __restrict__ out, int cols) {
  __shared__ float red[256];
  const int row = blockIdx.x, tid = threadIdx.x, bs = blockDim.x;
  const float* xr = x + (size_t)row * cols;
  float s = 0.f;
  for (int c = tid; c < cols; c += bs) s += xr[c];
  red[tid] = s; __syncthreads();
  for (int st = bs >> 1; st > 0; st >>= 1) {
    if (tid < st) red[tid] += red[tid + st];
    __syncthreads();
  }
  float mean = red[0] / (float)cols; __syncthreads();
  float vs = 0.f;
  for (int c = tid; c < cols; c += bs) { float d = xr[c] - mean; vs += d * d; }
  red[tid] = vs; __syncthreads();
  for (int st = bs >> 1; st > 0; st >>= 1) {
    if (tid < st) red[tid] += red[tid + st];
    __syncthreads();
  }
  float rstd = rsqrtf(red[0] / (float)cols + 1e-5f);
  for (int c = tid; c < cols; c += bs)
    out[(size_t)row * cols + c] = (xr[c] - mean) * rstd * g[c] + b[c];
}

// ------------------- encoder attention (key-padding mask) -------------------
__global__ void enc_attn_k(const float* __restrict__ q,
                           const float* __restrict__ k,
                           const float* __restrict__ v,
                           const int* __restrict__ item_seq,
                           float* __restrict__ out) {
  int gid = blockIdx.x * blockDim.x + threadIdx.x;
  if (gid >= B_ * EH_ * SI_) return;
  int qi = gid % SI_;
  int h  = (gid / SI_) % EH_;
  int b  = gid / (SI_ * EH_);
  float qr[EDH_];
  size_t qb = ((size_t)b * SI_ + qi) * HID_ + h * EDH_;
  for (int d = 0; d < EDH_; ++d) qr[d] = q[qb + d];
  float s[SI_];
  float m = -1e30f;
  for (int kk = 0; kk < SI_; ++kk) {
    size_t kb = ((size_t)b * SI_ + kk) * HID_ + h * EDH_;
    float sc = 0.f;
    for (int d = 0; d < EDH_; ++d) sc += qr[d] * k[kb + d];
    sc *= 0.17677669529663687f;              // 32^-0.5
    if (item_seq[b * SI_ + kk] == 0) sc = -1e9f;
    s[kk] = sc;
    m = fmaxf(m, sc);
  }
  float l = 0.f;
  for (int kk = 0; kk < SI_; ++kk) { s[kk] = __expf(s[kk] - m); l += s[kk]; }
  float inv = 1.f / l;
  float acc[EDH_];
  for (int d = 0; d < EDH_; ++d) acc[d] = 0.f;
  for (int kk = 0; kk < SI_; ++kk) {
    size_t kb = ((size_t)b * SI_ + kk) * HID_ + h * EDH_;
    float p = s[kk] * inv;
    for (int d = 0; d < EDH_; ++d) acc[d] += p * v[kb + d];
  }
  for (int d = 0; d < EDH_; ++d) out[qb + d] = acc[d];
}

// ------------------- gating (last-item -> softmax over experts) -------------------
__global__ void gate_k(const float* __restrict__ x, const int* __restrict__ lens,
                       const float* __restrict__ W1, const float* __restrict__ b1,
                       const float* __restrict__ W2, const float* __restrict__ b2,
                       float* __restrict__ wout) {
  __shared__ float h[HID_];
  __shared__ float g1[HID_];
  __shared__ float g2[E_];
  int b = blockIdx.x, t = threadIdx.x;
  int last = lens[b] - 1;
  h[t] = x[((size_t)b * SI_ + last) * HID_ + t];
  __syncthreads();
  float a = b1[t];
  for (int d = 0; d < HID_; ++d) a += h[d] * W1[t * HID_ + d];
  g1[t] = gelu_f(a);
  __syncthreads();
  if (t < E_) {
    float a2 = b2[t];
    for (int d = 0; d < HID_; ++d) a2 += g1[d] * W2[t * HID_ + d];
    g2[t] = a2;
  }
  __syncthreads();
  if (t == 0) {
    float m = g2[0];
    for (int e = 1; e < E_; ++e) m = fmaxf(m, g2[e]);
    float l = 0.f;
    for (int e = 0; e < E_; ++e) { g2[e] = __expf(g2[e] - m); l += g2[e]; }
    for (int e = 0; e < E_; ++e) wout[b * E_ + e] = g2[e] / l;
  }
}

// ------------------- LoRA mixture -------------------
// t[s,e,r] = x[s,:] . A[e,r,:]
__global__ void lora_t_k(const float* __restrict__ x, const float* __restrict__ A,
                         float* __restrict__ t) {
  int gid = blockIdx.x * blockDim.x + threadIdx.x;
  if (gid >= BSL_ * E_ * R_) return;
  int r = gid % R_;
  int e = (gid / R_) % E_;
  int s = gid / (R_ * E_);
  const float* xr = x + (size_t)s * D_;
  const float* ar = A + ((size_t)e * R_ + r) * D_;
  float acc = 0.f;
  for (int d = 0; d < D_; ++d) acc += xr[d] * ar[d];
  t[gid] = acc;
}

// out[s,o] += SCALE * sum_e w[b,e] * sum_r t[s,e,r] * Bm[e,o,r]
__global__ void lora_out_k(const float* __restrict__ t, const float* __restrict__ Bm,
                           const float* __restrict__ w, float* __restrict__ out) {
  int gid = blockIdx.x * blockDim.x + threadIdx.x;
  if (gid >= BSL_ * D_) return;
  int o = gid % D_;
  int s = gid / D_;
  int b = s / SL_;
  float total = 0.f;
  for (int e = 0; e < E_; ++e) {
    const float* tr = t + ((size_t)s * E_ + e) * R_;
    const float* br = Bm + ((size_t)e * D_ + o) * R_;
    float a = 0.f;
    for (int r = 0; r < R_; ++r) a += tr[r] * br[r];
    total += w[b * E_ + e] * a;
  }
  out[gid] += LORA_SCALE_ * total;
}

// ------------------- LM causal attention (online softmax, 2-pass) -------------------
__global__ void lm_attn_k(const float* __restrict__ q, const float* __restrict__ k,
                          const float* __restrict__ v, const int* __restrict__ am,
                          float* __restrict__ out) {
  int gid = blockIdx.x * blockDim.x + threadIdx.x;
  if (gid >= B_ * NH_ * SL_) return;
  int qi = gid % SL_;
  int h  = (gid / SL_) % NH_;
  int b  = gid / (SL_ * NH_);
  float qr[DH_];
  size_t qb = ((size_t)b * SL_ + qi) * D_ + h * DH_;
  for (int d = 0; d < DH_; ++d) qr[d] = q[qb + d];
  float m = -1e30f, l = 0.f;
  for (int kk = 0; kk < SL_; ++kk) {
    float sc = -1e9f;
    if (kk <= qi && am[b * SL_ + kk] != 0) {
      size_t kb = ((size_t)b * SL_ + kk) * D_ + h * DH_;
      sc = 0.f;
      for (int d = 0; d < DH_; ++d) sc += qr[d] * k[kb + d];
      sc *= 0.125f;                          // 64^-0.5
    }
    float nm = fmaxf(m, sc);
    l = l * __expf(m - nm) + __expf(sc - nm);
    m = nm;
  }
  float acc[DH_];
  for (int d = 0; d < DH_; ++d) acc[d] = 0.f;
  for (int kk = 0; kk < SL_; ++kk) {
    float sc = -1e9f;
    size_t kb = ((size_t)b * SL_ + kk) * D_ + h * DH_;
    if (kk <= qi && am[b * SL_ + kk] != 0) {
      sc = 0.f;
      for (int d = 0; d < DH_; ++d) sc += qr[d] * k[kb + d];
      sc *= 0.125f;
    }
    float p = __expf(sc - m);
    for (int d = 0; d < DH_; ++d) acc[d] += p * v[kb + d];
  }
  float inv = 1.f / l;
  for (int d = 0; d < DH_; ++d) out[qb + d] = acc[d] * inv;
}

// ------------------- heads -------------------
__global__ void last_logits_k(const float* __restrict__ yln,
                              const float* __restrict__ lm_head,
                              float* __restrict__ logits) {
  int gid = blockIdx.x * blockDim.x + threadIdx.x;
  if (gid >= B_ * V_) return;
  int vtok = gid % V_;
  int b = gid / V_;
  const float* yr = yln + ((size_t)b * SL_ + SL_ - 1) * D_;
  const float* wr = lm_head + (size_t)vtok * D_;
  float acc = 0.f;
  for (int d = 0; d < D_; ++d) acc += yr[d] * wr[d];
  logits[gid] = acc;
}

// one block per (b, item): stream the 640MB item_out_emb once per batch row
__global__ void item_scores_k(const float* __restrict__ logits,
                              const float* __restrict__ items,
                              float* __restrict__ out) {
  __shared__ float red[256];
  int blk = blockIdx.x;                       // B*(NI+1)
  int i = blk % (NI_ + 1);
  int b = blk / (NI_ + 1);
  const float* lr = logits + (size_t)b * V_;
  const float* ir = items + (size_t)i * V_;
  float s = 0.f;
  for (int vtok = threadIdx.x; vtok < V_; vtok += 256) s += lr[vtok] * ir[vtok];
  red[threadIdx.x] = s; __syncthreads();
  for (int st = 128; st > 0; st >>= 1) {
    if (threadIdx.x < st) red[threadIdx.x] += red[threadIdx.x + st];
    __syncthreads();
  }
  if (threadIdx.x == 0) out[blk] = red[0];
}

// =====================================================================
//                              host launcher
// =====================================================================
static inline int cdiv(int a, int b) { return (a + b - 1) / b; }
static inline int gemm_blocks128(int M, int N) { return (M / 128) * (N / 128); }

extern "C" void kernel_launch(void* const* d_in, const int* in_sizes, int n_in,
                              void* d_out, int out_size, void* d_ws, size_t ws_size,
                              hipStream_t stream) {
  (void)in_sizes; (void)n_in; (void)out_size; (void)ws_size;
  auto F = [&](int i) { return (const float*)d_in[i]; };
  const int* item_seq = (const int*)d_in[0];
  const int* item_len = (const int*)d_in[1];
  const int* llm_ids  = (const int*)d_in[2];
  const int* llm_am   = (const int*)d_in[3];
  // params (setup_inputs dict order)
  const float* enc_emb = F(4);
  const float *e_Wq = F(5),  *e_bq = F(6);
  const float *e_Wk = F(7),  *e_bk = F(8);
  const float *e_Wv = F(9),  *e_bv = F(10);
  const float *e_Wo = F(11), *e_bo = F(12);
  const float *e_ln1g = F(13), *e_ln1b = F(14);
  const float *e_W1 = F(15), *e_b1 = F(16);
  const float *e_W2 = F(17), *e_b2 = F(18);
  const float *e_ln2g = F(19), *e_ln2b = F(20);
  const float *g_W1 = F(21), *g_b1 = F(22);
  const float *g_W2 = F(23), *g_b2 = F(24);
  const float* tok_emb = F(25);
  const float *lnf_g = F(26), *lnf_b = F(27);
  const float* lm_head = F(28);
  const float* item_out = F(29);
  // layer params start at 30, 16 tensors per layer

  // ---- bump allocator over d_ws ----
  char* ws = (char*)d_ws;
  size_t off = 0;
  auto alloc = [&](size_t bytes) -> char* {
    char* p = ws + off;
    off += (bytes + 255) & ~(size_t)255;
    return p;
  };
  float* enc_x   = (float*)alloc((size_t)B_ * SI_ * HID_ * 4);
  float* enc_q   = (float*)alloc((size_t)B_ * SI_ * HID_ * 4);
  float* enc_kk  = (float*)alloc((size_t)B_ * SI_ * HID_ * 4);
  float* enc_vv  = (float*)alloc((size_t)B_ * SI_ * HID_ * 4);
  float* enc_a   = (float*)alloc((size_t)B_ * SI_ * HID_ * 4);
  float* enc_tmp = (float*)alloc((size_t)B_ * SI_ * HID_ * 4);
  float* enc_ff1 = (float*)alloc((size_t)B_ * SI_ * EFF_ * 4);
  float* wgate   = (float*)alloc((size_t)B_ * E_ * 4);
  float* y       = (float*)alloc((size_t)BSL_ * D_ * 4);
  float* h1      = (float*)alloc((size_t)BSL_ * D_ * 4);
  float* qh      = (float*)alloc((size_t)BSL_ * D_ * 4);
  float* kh      = (float*)alloc((size_t)BSL_ * D_ * 4);
  float* vh      = (float*)alloc((size_t)BSL_ * D_ * 4);
  float* attno   = (float*)alloc((size_t)BSL_ * D_ * 4);
  float* proj    = (float*)alloc((size_t)BSL_ * D_ * 4);
  float* m1      = (float*)alloc((size_t)BSL_ * DFF_ * 4);
  float* tbuf    = (float*)alloc((size_t)BSL_ * E_ * R_ * 4);
  _Float16* Xh   = (_Float16*)alloc((size_t)BSL_ * DFF_ * 2);  // max-K activations
  _Float16* Wh   = (_Float16*)alloc((size_t)DFF_ * D_ * 2);    // largest weight
  float* logits  = (float*)alloc((size_t)B_ * V_ * 4);

  const int T = 256;
  auto cast = [&](const float* s, _Float16* d, int n) {
    cast_f32_f16<<<cdiv(n, T), T, 0, stream>>>(s, d, n);
  };

  // ===================== sequence encoder =====================
  {
    int n = B_ * SI_ * HID_;
    gather_rows_k<<<cdiv(n, T), T, 0, stream>>>(enc_emb, item_seq, enc_x,
                                                B_ * SI_, HID_);
    int M = B_ * SI_;
    sgemm_f32_k<<<cdiv(M * HID_, T), T, 0, stream>>>(enc_x, e_Wq, e_bq, enc_q,
                                                     M, HID_, HID_, 0);
    sgemm_f32_k<<<cdiv(M * HID_, T), T, 0, stream>>>(enc_x, e_Wk, e_bk, enc_kk,
                                                     M, HID_, HID_, 0);
    sgemm_f32_k<<<cdiv(M * HID_, T), T, 0, stream>>>(enc_x, e_Wv, e_bv, enc_vv,
                                                     M, HID_, HID_, 0);
    enc_attn_k<<<cdiv(B_ * EH_ * SI_, 128), 128, 0, stream>>>(
        enc_q, enc_kk, enc_vv, item_seq, enc_a);
    sgemm_f32_k<<<cdiv(M * HID_, T), T, 0, stream>>>(enc_a, e_Wo, e_bo, enc_tmp,
                                                     M, HID_, HID_, 0);
    add_inplace_k<<<cdiv(n, T), T, 0, stream>>>(enc_x, enc_tmp, n);
    ln_rows_k<<<M, 128, 0, stream>>>(enc_x, e_ln1g, e_ln1b, enc_x, HID_);
    sgemm_f32_k<<<cdiv(M * EFF_, T), T, 0, stream>>>(enc_x, e_W1, e_b1, enc_ff1,
                                                     M, EFF_, HID_, 1 /*relu*/);
    sgemm_f32_k<<<cdiv(M * HID_, T), T, 0, stream>>>(enc_ff1, e_W2, e_b2,
                                                     enc_tmp, M, HID_, EFF_, 0);
    add_inplace_k<<<cdiv(n, T), T, 0, stream>>>(enc_x, enc_tmp, n);
    ln_rows_k<<<M, 128, 0, stream>>>(enc_x, e_ln2g, e_ln2b, enc_x, HID_);
    gate_k<<<B_, HID_, 0, stream>>>(enc_x, item_len, g_W1, g_b1, g_W2, g_b2,
                                    wgate);
  }

  // ===================== causal LM with mixed LoRA =====================
  {
    int nY = BSL_ * D_;
    gather_rows_k<<<cdiv(nY, T), T, 0, stream>>>(tok_emb, llm_ids, y, BSL_, D_);

    for (int L = 0; L < NL_; ++L) {
      int base = 30 + 16 * L;
      const float *ln1g = F(base + 0), *ln1b = F(base + 1);
      const float *Wq = F(base + 2), *Wk = F(base + 3);
      const float *Wv = F(base + 4), *Wo = F(base + 5);
      const float *Aq = F(base + 6), *Bq = F(base + 7);
      const float *Av = F(base + 8), *Bv = F(base + 9);
      const float *ln2g = F(base + 10), *ln2b = F(base + 11);
      const float *Wm1 = F(base + 12), *bm1 = F(base + 13);
      const float *Wm2 = F(base + 14), *bm2 = F(base + 15);

      // h1 = LN(y); q/k/v projections on the WMMA path
      ln_rows_k<<<BSL_, 256, 0, stream>>>(y, ln1g, ln1b, h1, D_);
      cast(h1, Xh, nY);
      cast(Wq, Wh, D_ * D_);
      wmma_gemm_nt<<<gemm_blocks128(BSL_, D_), 256, 0, stream>>>(Xh, Wh, qh, BSL_, D_, D_);
      cast(Wk, Wh, D_ * D_);
      wmma_gemm_nt<<<gemm_blocks128(BSL_, D_), 256, 0, stream>>>(Xh, Wh, kh, BSL_, D_, D_);
      cast(Wv, Wh, D_ * D_);
      wmma_gemm_nt<<<gemm_blocks128(BSL_, D_), 256, 0, stream>>>(Xh, Wh, vh, BSL_, D_, D_);

      // LoRA mixture adds (rank-16, f32)
      int nT = BSL_ * E_ * R_;
      lora_t_k<<<cdiv(nT, T), T, 0, stream>>>(h1, Aq, tbuf);
      lora_out_k<<<cdiv(nY, T), T, 0, stream>>>(tbuf, Bq, wgate, qh);
      lora_t_k<<<cdiv(nT, T), T, 0, stream>>>(h1, Av, tbuf);
      lora_out_k<<<cdiv(nY, T), T, 0, stream>>>(tbuf, Bv, wgate, vh);

      // causal attention + output projection + residual
      lm_attn_k<<<cdiv(B_ * NH_ * SL_, 128), 128, 0, stream>>>(qh, kh, vh,
                                                               llm_am, attno);
      cast(attno, Xh, nY);
      cast(Wo, Wh, D_ * D_);
      wmma_gemm_nt<<<gemm_blocks128(BSL_, D_), 256, 0, stream>>>(Xh, Wh, proj, BSL_, D_, D_);
      add_inplace_k<<<cdiv(nY, T), T, 0, stream>>>(y, proj, nY);

      // MLP: gelu(h2 @ Wm1^T + bm1) @ Wm2^T + bm2
      ln_rows_k<<<BSL_, 256, 0, stream>>>(y, ln2g, ln2b, h1, D_);
      cast(h1, Xh, nY);
      cast(Wm1, Wh, DFF_ * D_);
      wmma_gemm_nt<<<gemm_blocks128(BSL_, DFF_), 256, 0, stream>>>(Xh, Wh, m1, BSL_, DFF_, D_);
      bias_act_k<<<cdiv(BSL_ * DFF_, T), T, 0, stream>>>(m1, bm1, BSL_, DFF_, 2);
      cast(m1, Xh, BSL_ * DFF_);
      cast(Wm2, Wh, D_ * DFF_);
      wmma_gemm_nt<<<gemm_blocks128(BSL_, D_), 256, 0, stream>>>(Xh, Wh, proj, BSL_, D_, DFF_);
      bias_act_k<<<cdiv(nY, T), T, 0, stream>>>(proj, bm2, BSL_, D_, 0);
      add_inplace_k<<<cdiv(nY, T), T, 0, stream>>>(y, proj, nY);
    }

    // final LN, lm_head on last token, item scores
    ln_rows_k<<<BSL_, 256, 0, stream>>>(y, lnf_g, lnf_b, h1, D_);
    last_logits_k<<<cdiv(B_ * V_, T), T, 0, stream>>>(h1, lm_head, logits);
    item_scores_k<<<B_ * (NI_ + 1), 256, 0, stream>>>(logits, item_out,
                                                      (float*)d_out);
  }
}
